// DecoderRNN_82351702933950
// MI455X (gfx1250) — matrix operand order repaired and checked by reading, not compiled, not run on hardware
//
#include <hip/hip_runtime.h>
#include <hip/hip_bf16.h>
#include <math.h>

// ---------------- problem constants ----------------
#define B_      16384
#define S_      101
#define TSTEPS  100
#define V_      36
#define VP      48      // vocab padded to 3 WMMA n-tiles
#define E_      128
#define H_      256
#define ZD      128
#define MT      64      // batch rows per workgroup
#define NTHREADS 128    // 4 waves of 32, each wave owns 16 rows

typedef _Float16 v8h  __attribute__((ext_vector_type(8)));
typedef _Float16 v16h __attribute__((ext_vector_type(16)));
typedef float    v8f  __attribute__((ext_vector_type(8)));

// ---------------- workspace layout (bytes) ----------------
#define OFF_WZ     0u         // 512x128 f16      (131072 B)
#define OFF_WIH0   131072u    // 768x128 f16      (196608 B)
#define OFF_WHH0   327680u    // 768x256 f16      (393216 B)
#define OFF_WIH1   720896u    // 768x256 f16
#define OFF_WHH1   1114112u   // 768x256 f16
#define OFF_WOUT   1507328u   // 48x256  f16      (24576 B, rows 36..47 zero)
#define OFF_EMB    1531904u   // 36x128  f16      (9216 B)
#define OFF_MS     1541632u   // 16384   f32      (65536 B)

__device__ __forceinline__ v8f zero8() {
  v8f z = {0.f, 0.f, 0.f, 0.f, 0.f, 0.f, 0.f, 0.f};
  return z;
}

__device__ __forceinline__ v8f wmma_f16(v16h a, v16h b, v8f c) {
  // D = A(16x32) * B(32x16) + C(16x16 f32)
  return __builtin_amdgcn_wmma_f32_16x16x32_f16(
      /*neg_a=*/false, a, /*neg_b=*/false, b,
      /*c_mod=*/(short)0, c, /*reuse_a=*/false, /*reuse_b=*/false);
}

// A fragment (16x32 f16, rows striped per lane):
//   lane l<16 : row=l,     K [kk+0..15]
//   lane l>=16: row=l-16,  K [kk+16..31]
__device__ __forceinline__ v16h load_fragA(const _Float16* base, int kd,
                                           int row0, int kk, int lane) {
  const int half = lane >> 4;
  const int r    = lane & 15;
  const _Float16* p = base + (size_t)(row0 + r) * kd + kk + half * 16;
  v8h lo = *(const v8h*)(p);
  v8h hi = *(const v8h*)(p + 8);
  return __builtin_shufflevector(lo, hi, 0,1,2,3,4,5,6,7,8,9,10,11,12,13,14,15);
}

// B fragment (32x16 f16 = W^T tile; W row-major [N][K]):
//   lane l: column n = nbase + (l&15); 16 contiguous K values at kk + 16*(l>>4)
__device__ __forceinline__ v16h load_fragB(const _Float16* W, int kd,
                                           int nbase, int kk, int lane) {
  const int half = lane >> 4;
  const int n    = nbase + (lane & 15);
  const _Float16* p = W + (size_t)n * kd + kk + half * 16;
  v8h lo = *(const v8h*)(p);
  v8h hi = *(const v8h*)(p + 8);
  return __builtin_shufflevector(lo, hi, 0,1,2,3,4,5,6,7,8,9,10,11,12,13,14,15);
}

// Fused GRU layer for one wave's 16 rows. xin: [MT][KD_X] LDS f16,
// hin/hout: [MT][H_] LDS f16. Wih: [3H][KD_X], Whh: [3H][H_] global f16.
template <int KD_X>
__device__ __forceinline__ void gru_layer(const _Float16* xin,
                                          const _Float16* hin,
                                          _Float16* hout,
                                          const _Float16* __restrict__ Wih,
                                          const _Float16* __restrict__ Whh,
                                          const float* __restrict__ b_ih,
                                          const float* __restrict__ b_hh,
                                          int r0, int lane) {
  const int half = lane >> 4;
  const int lcol = lane & 15;
  for (int ct = 0; ct < H_ / 16; ++ct) {
    const int c = ct * 16;
    v8f ar = zero8(), az = zero8(), ain = zero8(), ahn = zero8();
    // input contribution: gi = x @ Wih^T
    for (int k = 0; k < KD_X; k += 32) {
      v16h a = load_fragA(xin, KD_X, r0, k, lane);
      ar  = wmma_f16(a, load_fragB(Wih, KD_X, c,        k, lane), ar);
      az  = wmma_f16(a, load_fragB(Wih, KD_X, H_ + c,   k, lane), az);
      ain = wmma_f16(a, load_fragB(Wih, KD_X, 2*H_ + c, k, lane), ain);
    }
    // hidden contribution: gh = h @ Whh^T (h_n kept separate for r-gating)
    for (int k = 0; k < H_; k += 32) {
      v16h a = load_fragA(hin, H_, r0, k, lane);
      ar  = wmma_f16(a, load_fragB(Whh, H_, c,        k, lane), ar);
      az  = wmma_f16(a, load_fragB(Whh, H_, H_ + c,   k, lane), az);
      ahn = wmma_f16(a, load_fragB(Whh, H_, 2*H_ + c, k, lane), ahn);
    }
    const int gc   = c + lcol;
    const float br  = b_ih[gc] + b_hh[gc];
    const float bzv = b_ih[H_ + gc] + b_hh[H_ + gc];
    const float bin = b_ih[2*H_ + gc];
    const float bhn = b_hh[2*H_ + gc];
#pragma unroll
    for (int v = 0; v < 8; ++v) {
      const int m = r0 + v + 8 * half;
      const float r  = 1.0f / (1.0f + __expf(-(ar[v] + br)));
      const float zg = 1.0f / (1.0f + __expf(-(az[v] + bzv)));
      const float nn = tanhf((ain[v] + bin) + r * (ahn[v] + bhn));
      const float ho = (float)hin[(size_t)m * H_ + gc];
      hout[(size_t)m * H_ + gc] = (_Float16)((1.0f - zg) * nn + zg * ho);
    }
  }
}

// ---------------- main persistent decode kernel ----------------
__global__ __launch_bounds__(NTHREADS, 1)
void decoder_rnn_kernel(const float* __restrict__ z, const int* __restrict__ x,
                        const float* __restrict__ bz,
                        const float* __restrict__ b_ih0, const float* __restrict__ b_hh0,
                        const float* __restrict__ b_ih1, const float* __restrict__ b_hh1,
                        const float* __restrict__ b_out,
                        const _Float16* __restrict__ wz16,
                        const _Float16* __restrict__ wih0, const _Float16* __restrict__ whh0,
                        const _Float16* __restrict__ wih1, const _Float16* __restrict__ whh1,
                        const _Float16* __restrict__ woutp,
                        const _Float16* __restrict__ emb16,
                        float* __restrict__ out, float* __restrict__ molsum) {
  __shared__ _Float16 sE[MT * E_];         // embeddings / z tile  (16 KB)
  __shared__ _Float16 sH0[2][MT * H_];     // layer-0 hidden, double buffered (64 KB)
  __shared__ _Float16 sH1[2][MT * H_];     // layer-1 hidden, double buffered (64 KB)
  __shared__ float    sLog[MT * VP];       // logits scratch (12 KB)
  __shared__ int      sTok[MT];

  const int tid   = threadIdx.x;
  const int lane  = tid & 31;
  const int wv    = tid >> 5;
  const int r0    = wv * 16;               // this wave's 16 local rows
  const int half  = lane >> 4;
  const int lcol  = lane & 15;
  const int gRow0 = blockIdx.x * MT;
  const int myrow = r0 + lcol;
  const int gb    = gRow0 + myrow;         // global batch row for this lane

  // ---- stage z tile into LDS as f16 ----
  for (int i = tid; i < MT * E_; i += NTHREADS)
    sE[i] = (_Float16)z[(size_t)gRow0 * ZD + i];
  __syncthreads();

  // ---- h init: relu(z @ Wz^T + bz) -> sH0[0], sH1[0] ----
  for (int ct = 0; ct < 2 * H_ / 16; ++ct) {
    const int c = ct * 16;
    v8f acc = zero8();
    for (int k = 0; k < ZD; k += 32) {
      v16h a = load_fragA(sE, ZD, r0, k, lane);
      acc = wmma_f16(a, load_fragB(wz16, ZD, c, k, lane), acc);
    }
    const int gc = c + lcol;
    const float bb = bz[gc];
#pragma unroll
    for (int v = 0; v < 8; ++v) {
      const int m = r0 + v + 8 * half;
      float hv = fmaxf(acc[v] + bb, 0.0f);
      if (gc < H_) sH0[0][(size_t)m * H_ + gc] = (_Float16)hv;
      else         sH1[0][(size_t)m * H_ + (gc - H_)] = (_Float16)hv;
    }
  }
  __syncthreads();

  // ---- token init (START=1), sequence lengths, first embedding ----
  if (lane < 16) sTok[myrow] = 1;
  float lossSum = 0.0f;
  float lenf = 1.0f;
  if (lane < 16) {
    int c = 0;
    for (int s = 0; s < S_; ++s) c += (x[(size_t)gb * S_ + s] != 0) ? 1 : 0;
    lenf = (float)c;
  }
  {
    const int tk = sTok[myrow];
    const _Float16* es = emb16 + (size_t)tk * E_ + half * 64;
    _Float16* ed = &sE[(size_t)myrow * E_ + half * 64];
#pragma unroll
    for (int j = 0; j < 8; ++j)
      *(v8h*)(ed + j * 8) = *(const v8h*)(es + j * 8);
  }

  // ---- 100 decode steps; waves are fully row-independent: NO barriers ----
  int cur = 0;
  for (int t = 0; t < TSTEPS; ++t) {
    const int nxt = cur ^ 1;
    gru_layer<E_>(sE,        sH0[cur], sH0[nxt], wih0, whh0, b_ih0, b_hh0, r0, lane);
    gru_layer<H_>(sH0[nxt],  sH1[cur], sH1[nxt], wih1, whh1, b_ih1, b_hh1, r0, lane);

    // ---- logits = h1 @ Wout^T + b_out (Wout padded to 48 rows) ----
    for (int ct = 0; ct < VP / 16; ++ct) {
      const int c = ct * 16;
      v8f acc = zero8();
      for (int k = 0; k < H_; k += 32) {
        v16h a = load_fragA(sH1[nxt], H_, r0, k, lane);
        acc = wmma_f16(a, load_fragB(woutp, H_, c, k, lane), acc);
      }
      const int gc = c + lcol;
      if (gc < V_) {
        const float bo = b_out[gc];
#pragma unroll
        for (int v = 0; v < 8; ++v) {
          const int m = r0 + v + 8 * half;
          sLog[(size_t)m * VP + gc] = acc[v] + bo;
        }
      }
    }

    // ---- log-softmax / nll / argmax: lanes 0..15, one row each ----
    if (lane < 16) {
      const float* lr = &sLog[(size_t)myrow * VP];
      float mx = -3.4e38f;
      for (int v = 0; v < V_; ++v) mx = fmaxf(mx, lr[v]);
      float se = 0.0f;
      for (int v = 0; v < V_; ++v) se += __expf(lr[v] - mx);
      const float lse = mx + __logf(se);
      float* op = out + ((size_t)gb * TSTEPS + t) * V_;
      const int tgt = x[(size_t)gb * S_ + (t + 1)];
      int amax = 0;
      float best = -3.4e38f;
      float nll = 0.0f;
      for (int v = 0; v < V_; ++v) {
        const float lp = lr[v] - lse;
        op[v] = lp;
        if (lp > best) { best = lp; amax = v; }
        if (v == tgt) nll = -lp;
      }
      if (tgt != 0) lossSum += nll;
      sTok[myrow] = amax;
    }

    // ---- gather next embedding (all 32 lanes; 128B per lane) ----
    {
      const int tk = sTok[myrow];
      const _Float16* es = emb16 + (size_t)tk * E_ + half * 64;
      _Float16* ed = &sE[(size_t)myrow * E_ + half * 64];
#pragma unroll
      for (int j = 0; j < 8; ++j)
        *(v8h*)(ed + j * 8) = *(const v8h*)(es + j * 8);
    }
    cur = nxt;
  }

  if (lane < 16) {
    out[(size_t)B_ * TSTEPS * V_ + gb] = lossSum / lenf;  // mol_loss
    molsum[gb] = lossSum;
  }
}

// ---------------- helper kernels ----------------
__global__ void cvt_f16_kernel(const float* __restrict__ s,
                               _Float16* __restrict__ d, int n) {
  int i = blockIdx.x * blockDim.x + threadIdx.x;
  if (i < n) d[i] = (_Float16)s[i];
}

__global__ void cvt_wout_kernel(const float* __restrict__ s,
                                _Float16* __restrict__ d) {
  int i = blockIdx.x * blockDim.x + threadIdx.x;
  if (i < VP * H_) {
    int row = i / H_;
    d[i] = (row < V_) ? (_Float16)s[i] : (_Float16)0.0f;
  }
}

__global__ void reduce_loss_kernel(const float* __restrict__ ms,
                                   float* __restrict__ out) {
  __shared__ float red[256];
  int tid = threadIdx.x;
  float s = 0.0f;
  for (int i = tid; i < B_; i += 256) s += ms[i];   // fixed order -> deterministic
  red[tid] = s;
  __syncthreads();
  for (int off = 128; off > 0; off >>= 1) {
    if (tid < off) red[tid] += red[tid + off];
    __syncthreads();
  }
  if (tid == 0) out[(size_t)B_ * TSTEPS * V_ + B_] = red[0] / (float)B_;
}

// ---------------- entry point ----------------
extern "C" void kernel_launch(void* const* d_in, const int* in_sizes, int n_in,
                              void* d_out, int out_size, void* d_ws, size_t ws_size,
                              hipStream_t stream) {
  (void)in_sizes; (void)n_in; (void)out_size; (void)ws_size;

  const float* z     = (const float*)d_in[0];
  const int*   x     = (const int*)d_in[1];
  const float* emb   = (const float*)d_in[2];
  const float* Wz    = (const float*)d_in[3];
  const float* bz    = (const float*)d_in[4];
  const float* W_ih0 = (const float*)d_in[5];
  const float* W_hh0 = (const float*)d_in[6];
  const float* b_ih0 = (const float*)d_in[7];
  const float* b_hh0 = (const float*)d_in[8];
  const float* W_ih1 = (const float*)d_in[9];
  const float* W_hh1 = (const float*)d_in[10];
  const float* b_ih1 = (const float*)d_in[11];
  const float* b_hh1 = (const float*)d_in[12];
  const float* W_out = (const float*)d_in[13];
  const float* b_out = (const float*)d_in[14];

  char* ws = (char*)d_ws;
  _Float16* wz16  = (_Float16*)(ws + OFF_WZ);
  _Float16* wih0  = (_Float16*)(ws + OFF_WIH0);
  _Float16* whh0  = (_Float16*)(ws + OFF_WHH0);
  _Float16* wih1  = (_Float16*)(ws + OFF_WIH1);
  _Float16* whh1  = (_Float16*)(ws + OFF_WHH1);
  _Float16* woutp = (_Float16*)(ws + OFF_WOUT);
  _Float16* emb16 = (_Float16*)(ws + OFF_EMB);
  float*    mols  = (float*)(ws + OFF_MS);
  float*    out   = (float*)d_out;

  const int T = 256;
  cvt_f16_kernel<<<(512 * ZD + T - 1) / T, T, 0, stream>>>(Wz, wz16, 512 * ZD);
  cvt_f16_kernel<<<(3 * H_ * E_ + T - 1) / T, T, 0, stream>>>(W_ih0, wih0, 3 * H_ * E_);
  cvt_f16_kernel<<<(3 * H_ * H_ + T - 1) / T, T, 0, stream>>>(W_hh0, whh0, 3 * H_ * H_);
  cvt_f16_kernel<<<(3 * H_ * H_ + T - 1) / T, T, 0, stream>>>(W_ih1, wih1, 3 * H_ * H_);
  cvt_f16_kernel<<<(3 * H_ * H_ + T - 1) / T, T, 0, stream>>>(W_hh1, whh1, 3 * H_ * H_);
  cvt_wout_kernel<<<(VP * H_ + T - 1) / T, T, 0, stream>>>(W_out, woutp);
  cvt_f16_kernel<<<(V_ * E_ + T - 1) / T, T, 0, stream>>>(emb, emb16, V_ * E_);

  decoder_rnn_kernel<<<B_ / MT, NTHREADS, 0, stream>>>(
      z, x, bz, b_ih0, b_hh0, b_ih1, b_hh1, b_out,
      wz16, wih0, whh0, wih1, whh1, woutp, emb16, out, mols);

  reduce_loss_kernel<<<1, 256, 0, stream>>>(mols, out);
}
